// GraphTransformerLayer_19885698580772
// MI455X (gfx1250) — matrix operand order repaired and checked by reading, not compiled, not run on hardware
//
#include <hip/hip_runtime.h>

// ---------------------------------------------------------------------------
// GraphTransformerLayer on gfx1250 (MI455X): WMMA f16->f32 GEMM pipeline.
// N=2048, D=256, H=8, head_dim=32, E=65536, ED=64, FFN=1024.
// All WMMA B operands are stored K-contiguous (transposed weights / Vt) so
// every fragment load is a pair of global_load_b128 per lane.
// ---------------------------------------------------------------------------

#define DEVI static __device__ __forceinline__

typedef __attribute__((ext_vector_type(16))) _Float16 v16h;
typedef __attribute__((ext_vector_type(8)))  _Float16 v8h;
typedef __attribute__((ext_vector_type(8)))  float    v8f;

// ---- WMMA fragment loaders (per CDNA5 ISA 7.12.2 layouts, wave32) ----------

// A-matrix 16x32 f16: lanes 0-15 -> row M=lane, K {0..7} then {16..23};
// lanes 16-31 -> row M=lane-16, K {8..15} then {24..31}.
DEVI v16h load_a_frag(const _Float16* __restrict__ A, int lda, int row0, int k0) {
  const int lane = threadIdx.x & 31;
  const _Float16* p = A + (size_t)(row0 + (lane & 15)) * lda + (k0 + ((lane >> 4) << 3));
  v8h lo = *(const v8h*)p;
  v8h hi = *(const v8h*)(p + 16);
  v16h a;
#pragma unroll
  for (int j = 0; j < 8; ++j) { a[j] = lo[j]; a[j + 8] = hi[j]; }
  return a;
}

// B-matrix 32x16 f16: lanes 0-15 -> col N=lane, K {0..7} then {16..23}; lanes 16-31 -> col
// N=lane-16, K {8..15} then {24..31} (matches A).  Operand stored K-contiguous:
// element (k, n) = Bt[n*ldk + k]  ->  two 16-byte loads per lane.
DEVI v16h load_b_frag_kc(const _Float16* __restrict__ Bt, int ldk, int k0, int col0) {
  const int lane = threadIdx.x & 31;
  const _Float16* p = Bt + (size_t)(col0 + (lane & 15)) * ldk + (k0 + ((lane >> 4) << 3));
  v8h lo = *(const v8h*)p;
  v8h hi = *(const v8h*)(p + 16);
  v16h b;
#pragma unroll
  for (int j = 0; j < 8; ++j) { b[j] = lo[j]; b[j + 8] = hi[j]; }
  return b;
}

DEVI v8f wmma_f16(v16h a, v16h b, v8f c) {
  return __builtin_amdgcn_wmma_f32_16x16x32_f16(false, a, false, b, (short)0, c,
                                                false, false);
}

DEVI float gelu_exact(float v) {
  return 0.5f * v * (1.0f + erff(v * 0.70710678118654752f));
}

// ---------------------------------------------------------------------------
// Generic WMMA GEMM: C[M,N] = epilogue(A[M,K] @ B[K,N]), B given transposed
// (Bt[N,K], K-contiguous).  block = 256 threads = 8 waves; each wave computes
// a 16x64 strip.  grid = (N/64, M/128).
// ---------------------------------------------------------------------------
template <bool BIAS, bool RES, bool GELU_ACT, bool OUTF16, bool TRANSOUT>
__global__ void __launch_bounds__(256)
gemm_wmma_kernel(const _Float16* __restrict__ A, const _Float16* __restrict__ Bt,
                 const float* __restrict__ bias, const float* __restrict__ res,
                 void* __restrict__ Cout, int M, int N, int K) {
  const int lane = threadIdx.x & 31;
  const int wave = threadIdx.x >> 5;
  const int row0 = (blockIdx.y * 8 + wave) * 16;
  const int col0 = blockIdx.x * 64;
  if (row0 >= M) return;

  v8f acc[4] = {};
  for (int k0 = 0; k0 < K; k0 += 32) {
    if (k0 + 32 < K) {  // gfx1250 global_prefetch_b8 of next K-chunk
      __builtin_prefetch(A + (size_t)(row0 + (lane & 15)) * K + k0 + 32, 0, 0);
      __builtin_prefetch(Bt + (size_t)(col0 + (lane & 15)) * K + k0 + 32, 0, 0);
    }
    v16h a = load_a_frag(A, K, row0, k0);
#pragma unroll
    for (int t = 0; t < 4; ++t) {
      v16h b = load_b_frag_kc(Bt, K, k0, col0 + t * 16);
      acc[t] = wmma_f16(a, b, acc[t]);
    }
  }

  const int rbase = row0 + ((lane >> 4) << 3);
#pragma unroll
  for (int t = 0; t < 4; ++t) {
    const int col = col0 + t * 16 + (lane & 15);
#pragma unroll
    for (int j = 0; j < 8; ++j) {
      const int row = rbase + j;
      float v = acc[t][j];
      if (BIAS) v += bias[col];
      if (GELU_ACT) v = gelu_exact(v);
      if (RES) v += res[(size_t)row * N + col];
      if (TRANSOUT) ((_Float16*)Cout)[(size_t)col * M + row] = (_Float16)v;
      else if (OUTF16) ((_Float16*)Cout)[(size_t)row * N + col] = (_Float16)v;
      else ((float*)Cout)[(size_t)row * N + col] = v;
    }
  }
}

// ---------------------------------------------------------------------------
// Per-head scores: S[n,m] = (Q_h[n,:] . K_h[m,:]) * scale.  head_dim=32 -> one
// v_wmma per 16x16 tile.  grid = (n/64, n/128), block 256.
// K-fragment is naturally K-contiguous (rows of K matrix).
// ---------------------------------------------------------------------------
__global__ void __launch_bounds__(256)
scores_wmma_kernel(const _Float16* __restrict__ Qh, const _Float16* __restrict__ Kh,
                   float* __restrict__ S, int n, int ld, float scale) {
  const int lane = threadIdx.x & 31;
  const int wave = threadIdx.x >> 5;
  const int row0 = (blockIdx.y * 8 + wave) * 16;
  const int col0 = blockIdx.x * 64;

  v16h a = load_a_frag(Qh, ld, row0, 0);
  const int rbase = row0 + ((lane >> 4) << 3);
#pragma unroll
  for (int t = 0; t < 4; ++t) {
    v16h b = load_b_frag_kc(Kh, ld, 0, col0 + t * 16);
    v8f acc = {};
    acc = wmma_f16(a, b, acc);
    const int col = col0 + t * 16 + (lane & 15);
#pragma unroll
    for (int j = 0; j < 8; ++j)
      S[(size_t)(rbase + j) * n + col] = acc[j] * scale;
  }
}

// ---------------------------------------------------------------------------
// Per-head attn @ V: AO[:, h*32+c] = P[N,N] @ V_h[N,32], with V stored
// transposed (Vt[c, n], ld = n) so B fragments are vector loads.  grid = n/128.
// ---------------------------------------------------------------------------
__global__ void __launch_bounds__(256)
av_wmma_kernel(const _Float16* __restrict__ P, const _Float16* __restrict__ Vt,
               _Float16* __restrict__ AO, int n, int D, int head) {
  const int lane = threadIdx.x & 31;
  const int wave = threadIdx.x >> 5;
  const int row0 = (blockIdx.x * 8 + wave) * 16;
  const _Float16* Vth = Vt + (size_t)(head * 32) * n;  // element (k,c) = Vth[c*n + k]

  v8f acc[2] = {};
  for (int k0 = 0; k0 < n; k0 += 32) {
    if (k0 + 32 < n)
      __builtin_prefetch(P + (size_t)(row0 + (lane & 15)) * n + k0 + 32, 0, 0);
    v16h a = load_a_frag(P, n, row0, k0);
#pragma unroll
    for (int t = 0; t < 2; ++t) {
      v16h b = load_b_frag_kc(Vth, n, k0, t * 16);
      acc[t] = wmma_f16(a, b, acc[t]);
    }
  }
  const int rbase = row0 + ((lane >> 4) << 3);
#pragma unroll
  for (int t = 0; t < 2; ++t) {
    const int col = head * 32 + t * 16 + (lane & 15);
#pragma unroll
    for (int j = 0; j < 8; ++j)
      AO[(size_t)(rbase + j) * D + col] = (_Float16)acc[t][j];
  }
}

// ---------------------------------------------------------------------------
// Masked row softmax over S row -> f16 P (0 where masked). One block per row.
// ---------------------------------------------------------------------------
__global__ void __launch_bounds__(256)
masked_softmax_kernel(const float* __restrict__ S, const unsigned* __restrict__ adj,
                      _Float16* __restrict__ P, int n) {
  const int row = blockIdx.x;
  const int tid = threadIdx.x;
  __shared__ float red[256];

  float vals[8];
  float mx = -3.4e38f;
#pragma unroll
  for (int i = 0; i < 8; ++i) {
    const int m = tid + i * 256;
    const unsigned bits = adj[(size_t)row * (n >> 5) + (m >> 5)];
    const bool ok = (bits >> (m & 31)) & 1u;
    const float v = ok ? S[(size_t)row * n + m] : -3.4e38f;
    vals[i] = v;
    mx = fmaxf(mx, v);
  }
  red[tid] = mx;
  __syncthreads();
  for (int s = 128; s > 0; s >>= 1) {
    if (tid < s) red[tid] = fmaxf(red[tid], red[tid + s]);
    __syncthreads();
  }
  mx = red[0];
  __syncthreads();

  float sum = 0.0f;
#pragma unroll
  for (int i = 0; i < 8; ++i) {
    const float e = (vals[i] > -1.0e37f) ? __expf(vals[i] - mx) : 0.0f;
    vals[i] = e;
    sum += e;
  }
  red[tid] = sum;
  __syncthreads();
  for (int s = 128; s > 0; s >>= 1) {
    if (tid < s) red[tid] += red[tid + s];
    __syncthreads();
  }
  const float inv = 1.0f / red[0];  // diag always present -> sum > 0
#pragma unroll
  for (int i = 0; i < 8; ++i)
    P[(size_t)row * n + tid + i * 256] = (_Float16)(vals[i] * inv);
}

// ---------------------------------------------------------------------------
// LayerNorm row kernel (D = 256 = blockDim) -> f16 output.
// ---------------------------------------------------------------------------
__global__ void __launch_bounds__(256)
layernorm_f16_kernel(const float* __restrict__ X, const float* __restrict__ g,
                     const float* __restrict__ b, _Float16* __restrict__ out, int D) {
  const int row = blockIdx.x;
  const int tid = threadIdx.x;
  __shared__ float red[256];

  const float v = X[(size_t)row * D + tid];
  red[tid] = v;
  __syncthreads();
  for (int s = 128; s > 0; s >>= 1) {
    if (tid < s) red[tid] += red[tid + s];
    __syncthreads();
  }
  const float mean = red[0] * (1.0f / 256.0f);
  __syncthreads();

  const float c = v - mean;
  red[tid] = c * c;
  __syncthreads();
  for (int s = 128; s > 0; s >>= 1) {
    if (tid < s) red[tid] += red[tid + s];
    __syncthreads();
  }
  const float var = red[0] * (1.0f / 256.0f);
  const float y = c * rsqrtf(var + 1e-5f) * g[tid] + b[tid];
  out[(size_t)row * D + tid] = (_Float16)y;
}

// ---------------------------------------------------------------------------
// Small helper kernels.
// ---------------------------------------------------------------------------

// in: W[K, N] f32 row-major; out: Wt[N, K] f16 row-major (K-contiguous).
__global__ void f32_to_f16_transpose_kernel(const float* __restrict__ in,
                                            _Float16* __restrict__ out, int K, int N) {
  const int i = blockIdx.x * 256 + threadIdx.x;
  if (i >= K * N) return;
  const int k = i / N;
  const int n = i - k * N;
  out[(size_t)n * K + k] = (_Float16)in[i];
}

__global__ void build_adj_kernel(const int* __restrict__ src, const int* __restrict__ dst,
                                 unsigned* __restrict__ adj, int E, int n) {
  const int e = blockIdx.x * 256 + threadIdx.x;
  if (e < E) {
    const int s = src[e], d = dst[e];
    atomicOr(&adj[(size_t)s * (n >> 5) + (d >> 5)], 1u << (d & 31));
  }
}

__global__ void set_diag_kernel(unsigned* __restrict__ adj, int n) {
  const int i = blockIdx.x * 256 + threadIdx.x;
  if (i < n) atomicOr(&adj[(size_t)i * (n >> 5) + (i >> 5)], 1u << (i & 31));
}

// edge_bias[e, h] = edge_attr[e,:] @ We + be  (ED=64, H=8; scalar VALU is fine)
__global__ void edge_bias_kernel(const float* __restrict__ ea, const float* __restrict__ We,
                                 const float* __restrict__ be, float* __restrict__ eb, int E) {
  const int e = blockIdx.x * 256 + threadIdx.x;
  if (e >= E) return;
  float acc[8];
#pragma unroll
  for (int h = 0; h < 8; ++h) acc[h] = be[h];
  for (int k = 0; k < 64; ++k) {
    const float a = ea[(size_t)e * 64 + k];
#pragma unroll
    for (int h = 0; h < 8; ++h) acc[h] += a * We[k * 8 + h];
  }
#pragma unroll
  for (int h = 0; h < 8; ++h) eb[(size_t)e * 8 + h] = acc[h];
}

__global__ void scatter_bias_kernel(const int* __restrict__ src, const int* __restrict__ dst,
                                    const float* __restrict__ eb, float* __restrict__ S,
                                    int E, int n, int h) {
  const int e = blockIdx.x * 256 + threadIdx.x;
  if (e < E) atomicAdd(&S[(size_t)src[e] * n + dst[e]], eb[(size_t)e * 8 + h]);
}

// ---------------------------------------------------------------------------
// Host-side orchestration.
// ---------------------------------------------------------------------------
extern "C" void kernel_launch(void* const* d_in, const int* in_sizes, int n_in,
                              void* d_out, int out_size, void* d_ws, size_t ws_size,
                              hipStream_t stream) {
  constexpr int Nn = 2048, Dd = 256, Hh = 8, Ee = 65536, Ff = 1024;
  const float scale = 0.17677669529663688f;  // 1/sqrt(32)

  const float* x   = (const float*)d_in[0];
  const int*   ei  = (const int*)d_in[1];
  const float* ea  = (const float*)d_in[2];
  const float* Wq  = (const float*)d_in[3];
  const float* bq  = (const float*)d_in[4];
  const float* Wk  = (const float*)d_in[5];
  const float* bk  = (const float*)d_in[6];
  const float* Wv  = (const float*)d_in[7];
  const float* bv  = (const float*)d_in[8];
  const float* Wo  = (const float*)d_in[9];
  const float* bo  = (const float*)d_in[10];
  const float* We  = (const float*)d_in[11];
  const float* be  = (const float*)d_in[12];
  const float* W1  = (const float*)d_in[13];
  const float* b1  = (const float*)d_in[14];
  const float* W2  = (const float*)d_in[15];
  const float* b2  = (const float*)d_in[16];
  const float* ln1g = (const float*)d_in[17];
  const float* ln1b = (const float*)d_in[18];
  const float* ln2g = (const float*)d_in[19];
  const float* ln2b = (const float*)d_in[20];
  const int* src = ei;
  const int* dst = ei + Ee;

  // Workspace layout (~41 MB total; 1 MB-granular, 256B aligned).
  char* base = (char*)d_ws;
  const size_t MB = 1u << 20;
  _Float16* hf16  = (_Float16*)(base + 0 * MB);        // [N,D]    1 MB
  _Float16* Qf16  = (_Float16*)(base + 1 * MB);        // [N,D]    1 MB
  _Float16* Kf16  = (_Float16*)(base + 2 * MB);        // [N,D]    1 MB
  _Float16* Vtf16 = (_Float16*)(base + 3 * MB);        // [D,N]    1 MB (transposed V)
  _Float16* AOf16 = (_Float16*)(base + 4 * MB);        // [N,D]    1 MB
  _Float16* h2f16 = (_Float16*)(base + 5 * MB);        // [N,D]    1 MB
  _Float16* Gf16  = (_Float16*)(base + 6 * MB);        // [N,4D]   4 MB
  _Float16* Wqt   = (_Float16*)(base + 10 * MB);               // 128 KB
  _Float16* Wkt   = (_Float16*)(base + 10 * MB + 128 * 1024);
  _Float16* Wvt   = (_Float16*)(base + 10 * MB + 256 * 1024);
  _Float16* Wot   = (_Float16*)(base + 10 * MB + 384 * 1024);
  _Float16* W1t   = (_Float16*)(base + 10 * MB + 512 * 1024);  // 512 KB
  _Float16* W2t   = (_Float16*)(base + 11 * MB);               // 512 KB
  float*    eb    = (float*)(base + 12 * MB);          // [E,H]    2 MB
  unsigned* adj   = (unsigned*)(base + 14 * MB);       // [N,N/32] 512 KB
  float*    X1    = (float*)(base + 15 * MB);          // [N,D]    2 MB
  float*    S     = (float*)(base + 17 * MB);          // [N,N]   16 MB
  _Float16* P     = (_Float16*)(base + 33 * MB);       // [N,N]    8 MB
  (void)ws_size; (void)in_sizes; (void)n_in; (void)out_size;

  // 1) f32 -> f16 weight conversion with transpose (Wt[n,k], K-contiguous).
  f32_to_f16_transpose_kernel<<<(Dd * Dd + 255) / 256, 256, 0, stream>>>(Wq, Wqt, Dd, Dd);
  f32_to_f16_transpose_kernel<<<(Dd * Dd + 255) / 256, 256, 0, stream>>>(Wk, Wkt, Dd, Dd);
  f32_to_f16_transpose_kernel<<<(Dd * Dd + 255) / 256, 256, 0, stream>>>(Wv, Wvt, Dd, Dd);
  f32_to_f16_transpose_kernel<<<(Dd * Dd + 255) / 256, 256, 0, stream>>>(Wo, Wot, Dd, Dd);
  f32_to_f16_transpose_kernel<<<(Dd * Ff + 255) / 256, 256, 0, stream>>>(W1, W1t, Dd, Ff);
  f32_to_f16_transpose_kernel<<<(Ff * Dd + 255) / 256, 256, 0, stream>>>(W2, W2t, Ff, Dd);

  // 2) LN1 -> h (f16).
  layernorm_f16_kernel<<<Nn, 256, 0, stream>>>(x, ln1g, ln1b, hf16, Dd);

  // 3) Q, K projections row-major; V projection written transposed (Vt).
  {
    dim3 grid(Dd / 64, Nn / 128);
    gemm_wmma_kernel<true, false, false, true, false><<<grid, 256, 0, stream>>>(
        hf16, Wqt, bq, nullptr, (void*)Qf16, Nn, Dd, Dd);
    gemm_wmma_kernel<true, false, false, true, false><<<grid, 256, 0, stream>>>(
        hf16, Wkt, bk, nullptr, (void*)Kf16, Nn, Dd, Dd);
    gemm_wmma_kernel<true, false, false, true, true><<<grid, 256, 0, stream>>>(
        hf16, Wvt, bv, nullptr, (void*)Vtf16, Nn, Dd, Dd);
  }

  // 4) Edge bias projection + adjacency bitmask (with self-loops).
  edge_bias_kernel<<<Ee / 256, 256, 0, stream>>>(ea, We, be, eb, Ee);
  hipMemsetAsync(adj, 0, (size_t)Nn * (Nn / 32) * sizeof(unsigned), stream);
  build_adj_kernel<<<Ee / 256, 256, 0, stream>>>(src, dst, adj, Ee, Nn);
  set_diag_kernel<<<Nn / 256, 256, 0, stream>>>(adj, Nn);

  // 5) Per-head attention (S and P reused across heads; both fit in 192MB L2).
  for (int h = 0; h < Hh; ++h) {
    dim3 sgrid(Nn / 64, Nn / 128);
    scores_wmma_kernel<<<sgrid, 256, 0, stream>>>(Qf16 + h * 32, Kf16 + h * 32,
                                                  S, Nn, Dd, scale);
    scatter_bias_kernel<<<Ee / 256, 256, 0, stream>>>(src, dst, eb, S, Ee, Nn, h);
    masked_softmax_kernel<<<Nn, 256, 0, stream>>>(S, adj, P, Nn);
    av_wmma_kernel<<<Nn / 128, 256, 0, stream>>>(P, Vtf16, AOf16, Nn, Dd, h);
  }

  // 6) Output projection + residual -> X1 (f32).
  {
    dim3 grid(Dd / 64, Nn / 128);
    gemm_wmma_kernel<true, true, false, false, false><<<grid, 256, 0, stream>>>(
        AOf16, Wot, bo, x, (void*)X1, Nn, Dd, Dd);
  }

  // 7) LN2 -> h2 (f16).
  layernorm_f16_kernel<<<Nn, 256, 0, stream>>>(X1, ln2g, ln2b, h2f16, Dd);

  // 8) FFN1 with exact GELU -> G (f16).
  {
    dim3 grid(Ff / 64, Nn / 128);
    gemm_wmma_kernel<true, false, true, true, false><<<grid, 256, 0, stream>>>(
        h2f16, W1t, b1, nullptr, (void*)Gf16, Nn, Ff, Dd);
  }

  // 9) FFN2 + residual -> d_out (f32).
  {
    dim3 grid(Dd / 64, Nn / 128);
    gemm_wmma_kernel<true, true, false, false, false><<<grid, 256, 0, stream>>>(
        Gf16, W2t, b2, X1, d_out, Nn, Dd, Ff);
  }
}
